// GAT_79156247265869
// MI455X (gfx1250) — matrix operand order repaired and checked by reading, not compiled, not run on hardware
//
#include <hip/hip_runtime.h>
#include <hip/hip_fp16.h>

typedef __attribute__((ext_vector_type(16))) _Float16 v16h;
typedef __attribute__((ext_vector_type(8)))  float    v8f;

#define LRELU_ALPHA 0.2f

// ---------------------------------------------------------------- weight packing
// w1: [H=8][F=512][O=64] f32 -> B1t[c][f] f16 (c = h*64+o), K-major for async staging
__global__ void pack_w1_kernel(const float* __restrict__ w1, _Float16* __restrict__ B1t) {
  int g = blockIdx.x * blockDim.x + threadIdx.x;   // 512*512 exact
  int f = g & 511;
  int c = g >> 9;
  int h = c >> 6, o = c & 63;
  B1t[(size_t)c * 512 + f] = (_Float16)w1[(size_t)h * (512 * 64) + (size_t)f * 64 + o];
}

// w2: [F=512][O=128] f32 -> B2t[o][f] f16
__global__ void pack_w2_kernel(const float* __restrict__ w2, _Float16* __restrict__ B2t) {
  int g = blockIdx.x * blockDim.x + threadIdx.x;   // 512*128 exact
  int f = g & 511;
  int o = g >> 9;
  B2t[(size_t)o * 512 + f] = (_Float16)w2[(size_t)f * 128 + o];
}

__global__ void cvt_f32_f16_kernel(const float* __restrict__ in, _Float16* __restrict__ out, int n) {
  int g = blockIdx.x * blockDim.x + threadIdx.x;
  if (g < n) out[g] = (_Float16)in[g];
}

// ---------------------------------------------------------------- WMMA GEMM
// C[N,M] = A[N,K] (f16, row-major) * Bt[M,K]^T (f16, K-major)
// 128x64 block tile, 8 waves (2x2 WMMA tiles each), double-buffered LDS filled by
// CDNA5 async global->LDS copies (ASYNCcnt), overlapping next-tile DMA with WMMA.
#define BM 128
#define BN 64
#define BK 32
#define LDT 40   // padded LDS pitch in halves (80B: 16B-aligned chunks, bank-skewed)

__global__ __launch_bounds__(256) void gemm_wmma_kernel(
    const _Float16* __restrict__ A, const _Float16* __restrict__ Bt,
    float* __restrict__ C, int K, int M) {
  __shared__ __align__(16) _Float16 sA[2][BM * LDT];
  __shared__ __align__(16) _Float16 sB[2][BN * LDT];
  const int tid  = threadIdx.x;
  const int wave = tid >> 5;
  const int lane = tid & 31;
  const int bm = blockIdx.y * BM;
  const int bn = blockIdx.x * BN;
  const int wr = (wave >> 1) << 5;   // 0,32,64,96
  const int wc = (wave & 1) << 5;    // 0,32
  const int lm   = lane & 15;
  const int khiA = (lane < 16) ? 0 : 8;    // ISA f16 A frag: K {0..7,16..23} / {8..15,24..31}
  const int khiB = (lane < 16) ? 0 : 8;    // ISA f16 B frag: K {0..7,16..23} / {8..15,24..31}

  // staging: 16-byte chunks. A tile = 128 rows x 64B -> 2 chunks/thread; B tile = 64 rows x 64B -> 1.
  const int rowA0 = tid >> 2;             // 0..63
  const int rowA1 = rowA0 + 64;           // 64..127
  const int chH   = (tid & 3) << 3;       // half-offset within row: 0,8,16,24
  const unsigned long long gA0 = (unsigned long long)(A + (size_t)(bm + rowA0) * K + chH);
  const unsigned long long gA1 = (unsigned long long)(A + (size_t)(bm + rowA1) * K + chH);
  const unsigned long long gB  = (unsigned long long)(Bt + (size_t)(bn + rowA0) * K + chH);
  unsigned lA0[2], lA1[2], lB[2];
#pragma unroll
  for (int b = 0; b < 2; ++b) {
    lA0[b] = (unsigned)(unsigned long long)&sA[b][rowA0 * LDT + chH];
    lA1[b] = (unsigned)(unsigned long long)&sA[b][rowA1 * LDT + chH];
    lB[b]  = (unsigned)(unsigned long long)&sB[b][rowA0 * LDT + chH];
  }

  v8f acc00 = {}, acc01 = {}, acc10 = {}, acc11 = {};
  const int nk = K / BK;

  // prologue: async-stage tile 0 into buffer 0
  asm volatile("global_load_async_to_lds_b128 %0, %1, off" :: "v"(lA0[0]), "v"(gA0) : "memory");
  asm volatile("global_load_async_to_lds_b128 %0, %1, off" :: "v"(lA1[0]), "v"(gA1) : "memory");
  asm volatile("global_load_async_to_lds_b128 %0, %1, off" :: "v"(lB[0]),  "v"(gB)  : "memory");

  for (int t = 0; t < nk; ++t) {
    const int buf = t & 1;
    __syncthreads();                       // WAR: all waves done reading buf^1 before overwrite
    if (t + 1 < nk) {
      const unsigned long long koff = (unsigned long long)((t + 1) * BK) * 2ull; // bytes
      const int nb = buf ^ 1;
      unsigned long long a0 = gA0 + koff, a1 = gA1 + koff, b0 = gB + koff;
      asm volatile("global_load_async_to_lds_b128 %0, %1, off" :: "v"(lA0[nb]), "v"(a0) : "memory");
      asm volatile("global_load_async_to_lds_b128 %0, %1, off" :: "v"(lA1[nb]), "v"(a1) : "memory");
      asm volatile("global_load_async_to_lds_b128 %0, %1, off" :: "v"(lB[nb]),  "v"(b0) : "memory");
      asm volatile("s_wait_asynccnt 0x3" ::: "memory");   // tile-t copies done; tile-t+1 in flight
    } else {
      asm volatile("s_wait_asynccnt 0x0" ::: "memory");
    }
    __syncthreads();                       // RAW: every wave's tile-t copies visible

    v16h a0v, a1v, b0v, b1v;
    const _Float16* pa0 = &sA[buf][(wr + lm) * LDT + khiA];
    ((uint4*)&a0v)[0] = *(const uint4*)(pa0);
    ((uint4*)&a0v)[1] = *(const uint4*)(pa0 + 16);
    const _Float16* pa1 = &sA[buf][(wr + 16 + lm) * LDT + khiA];
    ((uint4*)&a1v)[0] = *(const uint4*)(pa1);
    ((uint4*)&a1v)[1] = *(const uint4*)(pa1 + 16);
    const _Float16* pb0 = &sB[buf][(wc + lm) * LDT + khiB];
    ((uint4*)&b0v)[0] = *(const uint4*)(pb0);
    ((uint4*)&b0v)[1] = *(const uint4*)(pb0 + 16);
    const _Float16* pb1 = &sB[buf][(wc + 16 + lm) * LDT + khiB];
    ((uint4*)&b1v)[0] = *(const uint4*)(pb1);
    ((uint4*)&b1v)[1] = *(const uint4*)(pb1 + 16);

    acc00 = __builtin_amdgcn_wmma_f32_16x16x32_f16(false, a0v, false, b0v, (short)0, acc00, false, false);
    acc01 = __builtin_amdgcn_wmma_f32_16x16x32_f16(false, a0v, false, b1v, (short)0, acc01, false, false);
    acc10 = __builtin_amdgcn_wmma_f32_16x16x32_f16(false, a1v, false, b0v, (short)0, acc10, false, false);
    acc11 = __builtin_amdgcn_wmma_f32_16x16x32_f16(false, a1v, false, b1v, (short)0, acc11, false, false);
  }

  // C/D layout: reg r -> row r (lanes 0-15) / r+8 (lanes 16-31); col = lane&15
  const int rhalf = (lane < 16) ? 0 : 8;
  const int col0 = bn + wc + lm;
  const int row0 = bm + wr + rhalf;
#pragma unroll
  for (int r = 0; r < 8; ++r) {
    C[(size_t)(row0 + r) * M + col0]            = acc00[r];
    C[(size_t)(row0 + r) * M + col0 + 16]       = acc01[r];
    C[(size_t)(row0 + 16 + r) * M + col0]       = acc10[r];
    C[(size_t)(row0 + 16 + r) * M + col0 + 16]  = acc11[r];
  }
}

// ---------------------------------------------------------------- attention projections
__global__ void alpha_kernel(const float* __restrict__ Hm, const float* __restrict__ att,
                             float* __restrict__ as_, float* __restrict__ at_,
                             int N, int H, int O) {
  int g = blockIdx.x * blockDim.x + threadIdx.x;
  if (g >= N * H) return;
  int h = g % H, n = g / H;
  const float* hp = Hm + (size_t)n * (H * O) + h * O;
  const float* a0 = att + (size_t)h * 2 * O;
  const float* a1 = a0 + O;
  float s0 = 0.f, s1 = 0.f;
  for (int o = 0; o < O; ++o) { float v = hp[o]; s0 += v * a0[o]; s1 += v * a1[o]; }
  as_[g] = s0; at_[g] = s1;
}

// per-edge logits + per-column running max (values >= 0 -> uint-bit atomicMax is monotone)
__global__ void edge_logits_kernel(const int* __restrict__ el, const float* __restrict__ as_,
                                   const float* __restrict__ at_, float* __restrict__ eArr,
                                   unsigned int* __restrict__ colmax, int E, int H) {
  int g = blockIdx.x * blockDim.x + threadIdx.x;
  if (g >= E * H) return;
  int h = g % H, ed = g / H;
  int src = el[ed], tgt = el[E + ed];
  float e = as_[src * H + h] + at_[tgt * H + h];
  e = (e > 0.f) ? e : (LRELU_ALPHA * e);
  eArr[g] = e;
  atomicMax(colmax + tgt * H + h, __float_as_uint(fmaxf(e, 0.f)));
}

__global__ void colsum_kernel(const int* __restrict__ el, const float* __restrict__ eArr,
                              const unsigned int* __restrict__ colmax,
                              float* __restrict__ colsum, int* __restrict__ cnt, int E, int H) {
  int g = blockIdx.x * blockDim.x + threadIdx.x;
  if (g >= E * H) return;
  int h = g % H, ed = g / H;
  int tgt = el[E + ed];
  float m = __uint_as_float(colmax[tgt * H + h]);
  atomicAdd(colsum + tgt * H + h, __expf(eArr[g] - m));
  if (h == 0) atomicAdd(cnt + tgt, 1);
}

__global__ void coeff_kernel(const unsigned int* __restrict__ colmax, const float* __restrict__ colsum,
                             const int* __restrict__ cnt, float* __restrict__ cinv,
                             float* __restrict__ rD, int N, int H) {
  int g = blockIdx.x * blockDim.x + threadIdx.x;
  if (g >= N * H) return;
  int j = g / H;
  float m  = __uint_as_float(colmax[g]);
  float em = __expf(-m);
  float D  = colsum[g] + (float)(N - cnt[j]) * em;
  float r  = 1.0f / D;
  cinv[g] = em * r;
  rD[g]   = r;
}

__global__ void edgew_kernel(const int* __restrict__ el, const float* __restrict__ eArr,
                             const unsigned int* __restrict__ colmax, const float* __restrict__ cinv,
                             const float* __restrict__ rD, float* __restrict__ wArr, int E, int H) {
  int g = blockIdx.x * blockDim.x + threadIdx.x;
  if (g >= E * H) return;
  int h = g % H, ed = g / H;
  int cj = el[E + ed] * H + h;
  float m = __uint_as_float(colmax[cj]);
  wArr[g] = __expf(eArr[g] - m) * rD[cj] - cinv[cj];
}

// rank-1 term: base[c] = sum_j cinv[j,h] * Hm[j,c]
__global__ void base_kernel(const float* __restrict__ Hm, const float* __restrict__ cinv,
                            float* __restrict__ base, int N, int H, int O) {
  int c = blockIdx.x;
  int h = c / O;
  __shared__ float red[256];
  float s = 0.f;
  for (int j = threadIdx.x; j < N; j += 256)
    s += cinv[j * H + h] * Hm[(size_t)j * (H * O) + c];
  red[threadIdx.x] = s;
  __syncthreads();
  for (int w = 128; w > 0; w >>= 1) {
    if (threadIdx.x < w) red[threadIdx.x] += red[threadIdx.x + w];
    __syncthreads();
  }
  if (threadIdx.x == 0) base[c] = red[0];
}

__global__ void initout_kernel(const float* __restrict__ base, float* __restrict__ out, int total, int HO) {
  int g = blockIdx.x * blockDim.x + threadIdx.x;
  if (g < total) out[g] = base[g % HO];
}

// sparse part: out[src, h*O+o] += w[edge,h] * Hm[tgt, h*O+o]
__global__ void scatter_kernel(const int* __restrict__ el, const float* __restrict__ wArr,
                               const float* __restrict__ Hm, float* __restrict__ out,
                               int E, int H, int O, int lgO, int lgH, int HO) {
  int g = blockIdx.x * blockDim.x + threadIdx.x;  // < E*H*O
  int o  = g & (O - 1);
  int eh = g >> lgO;
  int h  = eh & (H - 1);
  int ed = eh >> lgH;
  float w = wArr[eh];
  int tgt = el[E + ed];
  int src = el[ed];
  float hv = Hm[(size_t)tgt * HO + (h << lgO) + o];
  atomicAdd(out + (size_t)src * HO + (h << lgO) + o, w * hv);
}

// elu(out + bias) -> f16 (input of next WMMA GEMM)
__global__ void elubias_f16_kernel(const float* __restrict__ in, const float* __restrict__ b,
                                   _Float16* __restrict__ out, int total, int C) {
  int g = blockIdx.x * blockDim.x + threadIdx.x;
  if (g >= total) return;
  float v = in[g] + b[g % C];
  out[g] = (_Float16)(v > 0.f ? v : expm1f(v));
}

// final: log_softmax over 128 features per node
__global__ void logsoftmax_kernel(const float* __restrict__ in, const float* __restrict__ b,
                                  float* __restrict__ out) {
  int n = blockIdx.x, t = threadIdx.x;
  __shared__ float red[128];
  float v = in[n * 128 + t] + b[t];
  red[t] = v; __syncthreads();
  for (int w = 64; w > 0; w >>= 1) { if (t < w) red[t] = fmaxf(red[t], red[t + w]); __syncthreads(); }
  float m = red[0]; __syncthreads();
  red[t] = __expf(v - m); __syncthreads();
  for (int w = 64; w > 0; w >>= 1) { if (t < w) red[t] += red[t + w]; __syncthreads(); }
  float ls = logf(red[0]);
  out[n * 128 + t] = v - m - ls;
}

// ---------------------------------------------------------------- launcher
extern "C" void kernel_launch(void* const* d_in, const int* in_sizes, int n_in,
                              void* d_out, int out_size, void* d_ws, size_t ws_size,
                              hipStream_t stream) {
  const float* x    = (const float*)d_in[0];
  const int*   el   = (const int*)d_in[1];
  const float* w1   = (const float*)d_in[2];
  const float* att1 = (const float*)d_in[3];
  const float* b1   = (const float*)d_in[4];
  const float* w2   = (const float*)d_in[5];
  const float* att2 = (const float*)d_in[6];
  const float* b2   = (const float*)d_in[7];
  float* out = (float*)d_out;

  const int N = 4096, E = 131072, F = 512;
  const int H1 = 8, O1 = 64, HO1 = 512;
  const int H2 = 1, O2 = 128, HO2 = 128;

  char* ws = (char*)d_ws;
  size_t off = 0;
  auto alloc = [&](size_t bytes) -> void* {
    void* p = ws + off;
    off = (off + bytes + 255) & ~(size_t)255;
    return p;
  };

  _Float16* X16  = (_Float16*)alloc((size_t)N * F * 2);
  _Float16* B1t  = (_Float16*)alloc((size_t)F * HO1 * 2);
  _Float16* B2t  = (_Float16*)alloc((size_t)HO1 * HO2 * 2);
  float* H1m     = (float*)alloc((size_t)N * HO1 * 4);
  float* AS1     = (float*)alloc((size_t)N * H1 * 4);
  float* AT1     = (float*)alloc((size_t)N * H1 * 4);
  float* E1      = (float*)alloc((size_t)E * H1 * 4);
  float* W1e     = (float*)alloc((size_t)E * H1 * 4);
  unsigned* CM1  = (unsigned*)alloc((size_t)N * H1 * 4);
  float* CS1     = (float*)alloc((size_t)N * H1 * 4);
  int*   CNT1    = (int*)alloc((size_t)N * 4);
  float* CI1     = (float*)alloc((size_t)N * H1 * 4);
  float* RD1     = (float*)alloc((size_t)N * H1 * 4);
  float* BASE1   = (float*)alloc((size_t)HO1 * 4);
  float* OUT1    = (float*)alloc((size_t)N * HO1 * 4);
  _Float16* HIN2 = (_Float16*)alloc((size_t)N * HO1 * 2);
  float* H2m     = (float*)alloc((size_t)N * HO2 * 4);
  float* AS2     = (float*)alloc((size_t)N * H2 * 4);
  float* AT2     = (float*)alloc((size_t)N * H2 * 4);
  float* E2      = (float*)alloc((size_t)E * H2 * 4);
  float* W2e     = (float*)alloc((size_t)E * H2 * 4);
  unsigned* CM2  = (unsigned*)alloc((size_t)N * 4);
  float* CS2     = (float*)alloc((size_t)N * 4);
  int*   CNT2    = (int*)alloc((size_t)N * 4);
  float* CI2     = (float*)alloc((size_t)N * 4);
  float* RD2     = (float*)alloc((size_t)N * 4);
  float* BASE2   = (float*)alloc((size_t)HO2 * 4);
  float* OUT2    = (float*)alloc((size_t)N * HO2 * 4);

  hipMemsetAsync(CM1, 0, (size_t)N * H1 * 4, stream);
  hipMemsetAsync(CS1, 0, (size_t)N * H1 * 4, stream);
  hipMemsetAsync(CNT1, 0, (size_t)N * 4, stream);
  hipMemsetAsync(CM2, 0, (size_t)N * 4, stream);
  hipMemsetAsync(CS2, 0, (size_t)N * 4, stream);
  hipMemsetAsync(CNT2, 0, (size_t)N * 4, stream);

  cvt_f32_f16_kernel<<<(N * F) / 256, 256, 0, stream>>>(x, X16, N * F);
  pack_w1_kernel<<<(F * HO1) / 256, 256, 0, stream>>>(w1, B1t);
  pack_w2_kernel<<<(HO1 * HO2) / 256, 256, 0, stream>>>(w2, B2t);

  // ---- layer 1
  gemm_wmma_kernel<<<dim3(HO1 / 64, N / 128), 256, 0, stream>>>(X16, B1t, H1m, F, HO1);
  alpha_kernel<<<(N * H1 + 255) / 256, 256, 0, stream>>>(H1m, att1, AS1, AT1, N, H1, O1);
  edge_logits_kernel<<<(E * H1) / 256, 256, 0, stream>>>(el, AS1, AT1, E1, CM1, E, H1);
  colsum_kernel<<<(E * H1) / 256, 256, 0, stream>>>(el, E1, CM1, CS1, CNT1, E, H1);
  coeff_kernel<<<(N * H1 + 255) / 256, 256, 0, stream>>>(CM1, CS1, CNT1, CI1, RD1, N, H1);
  edgew_kernel<<<(E * H1) / 256, 256, 0, stream>>>(el, E1, CM1, CI1, RD1, W1e, E, H1);
  base_kernel<<<HO1, 256, 0, stream>>>(H1m, CI1, BASE1, N, H1, O1);
  initout_kernel<<<(N * HO1) / 256, 256, 0, stream>>>(BASE1, OUT1, N * HO1, HO1);
  scatter_kernel<<<(E * H1 * O1) / 256, 256, 0, stream>>>(el, W1e, H1m, OUT1, E, H1, O1, 6, 3, HO1);
  elubias_f16_kernel<<<(N * HO1) / 256, 256, 0, stream>>>(OUT1, b1, HIN2, N * HO1, HO1);

  // ---- layer 2
  gemm_wmma_kernel<<<dim3(HO2 / 64, N / 128), 256, 0, stream>>>(HIN2, B2t, H2m, HO1, HO2);
  alpha_kernel<<<(N * H2 + 255) / 256, 256, 0, stream>>>(H2m, att2, AS2, AT2, N, H2, O2);
  edge_logits_kernel<<<(E * H2) / 256, 256, 0, stream>>>(el, AS2, AT2, E2, CM2, E, H2);
  colsum_kernel<<<(E * H2) / 256, 256, 0, stream>>>(el, E2, CM2, CS2, CNT2, E, H2);
  coeff_kernel<<<(N * H2 + 255) / 256, 256, 0, stream>>>(CM2, CS2, CNT2, CI2, RD2, N, H2);
  edgew_kernel<<<(E * H2) / 256, 256, 0, stream>>>(el, E2, CM2, CI2, RD2, W2e, E, H2);
  base_kernel<<<HO2, 256, 0, stream>>>(H2m, CI2, BASE2, N, H2, O2);
  initout_kernel<<<(N * HO2) / 256, 256, 0, stream>>>(BASE2, OUT2, N * HO2, HO2);
  scatter_kernel<<<(E * H2 * O2) / 256, 256, 0, stream>>>(el, W2e, H2m, OUT2, E, H2, O2, 7, 0, HO2);
  logsoftmax_kernel<<<N, 128, 0, stream>>>(OUT2, b2, out);
}
